// Attention_30365418783011
// MI455X (gfx1250) — matrix operand order repaired and checked by reading, not compiled, not run on hardware
//
#include <hip/hip_runtime.h>
#include <stdint.h>

#define S_LEN 4096
#define D_DIM 256
#define B_DIM 4
#define R_ROWS (B_DIM * S_LEN) // 16384 rows total

typedef __attribute__((ext_vector_type(16))) __bf16 v16bf;
typedef __attribute__((ext_vector_type(8)))  float  v8f;
typedef __attribute__((ext_vector_type(4)))  int    v4i;

static_assert(sizeof(v16bf) == 32, "v16bf must be 32 bytes");

union ABfrag { uint4 u[2]; v16bf v; };

__device__ __forceinline__ unsigned short f2bf(float f) {
    unsigned u = __float_as_uint(f);
    u += 0x7fffu + ((u >> 16) & 1u);   // round-to-nearest-even
    return (unsigned short)(u >> 16);
}

// ---- CDNA5 async global->LDS copy (ASYNCcnt-tracked), 16B per lane ----
__device__ __forceinline__ void async_copy16(const unsigned short* g,
                                             unsigned short* l) {
#if __has_builtin(__builtin_amdgcn_global_load_async_to_lds_b128)
    __builtin_amdgcn_global_load_async_to_lds_b128(
        (__attribute__((address_space(1))) v4i*)g,
        (__attribute__((address_space(3))) v4i*)l, 0, 0);
#else
    unsigned laddr = (unsigned)(unsigned long long)
        (__attribute__((address_space(3))) unsigned short*)l;
    asm volatile("global_load_async_to_lds_b128 %0, %1, off"
                 :: "v"(laddr), "v"(g) : "memory");
#endif
}

__device__ __forceinline__ void wait_async0() {
#if __has_builtin(__builtin_amdgcn_s_wait_asynccnt)
    __builtin_amdgcn_s_wait_asynccnt(0);
#else
    asm volatile("s_wait_asynccnt 0" ::: "memory");
#endif
}

__device__ __forceinline__ void lds_fence() {
    asm volatile("s_wait_dscnt 0" ::: "memory");
}

// ---------------------------------------------------------------- conversion
__global__ void cvt_bf16_kernel(const float* __restrict__ in,
                                unsigned short* __restrict__ out, int n) {
    for (int i = blockIdx.x * blockDim.x + threadIdx.x; i < n;
         i += gridDim.x * blockDim.x)
        out[i] = f2bf(in[i]);
}

// ---------------------------------------------------------------- row mask
__global__ void mask_kernel(const unsigned short* __restrict__ Qb,
                            const unsigned short* __restrict__ Kb,
                            float* __restrict__ m, int n) {
    for (int i = blockIdx.x * blockDim.x + threadIdx.x; i < n;
         i += gridDim.x * blockDim.x) {
        unsigned short q = Qb[(size_t)i * D_DIM];
        unsigned short k = Kb[(size_t)i * D_DIM];
        m[i] = (((q & 0x7fff) != 0) && ((k & 0x7fff) != 0)) ? 1.0f : 0.0f;
    }
}

// ---------------------------------------------------------------- projection
// Y = X(bf16, Rx256) @ W(bf16, 256x256 row-major [n][k])^T + bias
// MODE 0: bf16 row-major out. MODE 1: bf16 transposed (Vt[b][d][s]).
// MODE 2: fp32 row-major out.
template <int MODE>
__global__ __launch_bounds__(128) void proj_kernel(
    const unsigned short* __restrict__ X,
    const unsigned short* __restrict__ W,
    const float* __restrict__ bias,
    void* __restrict__ outp)
{
    const int wave = threadIdx.x >> 5;
    const int t    = threadIdx.x & 31;
    const int half = t >> 4;
    const int ln   = t & 15;
    const int row0 = blockIdx.x * 16;
    const int n0g  = wave * 64;

    const v8f zerov = {0.f,0.f,0.f,0.f,0.f,0.f,0.f,0.f};
    v8f acc[4];
#pragma unroll
    for (int i = 0; i < 4; i++) acc[i] = zerov;

    const int arow = row0 + ln;
#pragma unroll
    for (int kk = 0; kk < 8; kk++) {
        const int kb = kk * 32;
        ABfrag a;
        a.u[0] = *(const uint4*)(X + (size_t)arow * D_DIM + kb + 16 * half);
        a.u[1] = *(const uint4*)(X + (size_t)arow * D_DIM + kb + 16 * half + 8);
#pragma unroll
        for (int nt = 0; nt < 4; nt++) {
            const int n = n0g + nt * 16 + ln;
            ABfrag b;
            const uint4* bp = (const uint4*)(W + (size_t)n * D_DIM + kb + 16 * half);
            b.u[0] = bp[0];
            b.u[1] = bp[1];
            acc[nt] = __builtin_amdgcn_wmma_f32_16x16x32_bf16(
                false, a.v, false, b.v, (short)0, acc[nt], false, false);
        }
    }

#pragma unroll
    for (int nt = 0; nt < 4; nt++) {
        const int col = n0g + nt * 16 + ln;
        const float bval = bias[col];
#pragma unroll
        for (int r = 0; r < 8; r++) {
            const int row = row0 + r + 8 * half;
            const float val = acc[nt][r] + bval;
            if (MODE == 0) {
                ((unsigned short*)outp)[(size_t)row * D_DIM + col] = f2bf(val);
            } else if (MODE == 1) {
                const int bb = row / S_LEN, ss = row % S_LEN;
                ((unsigned short*)outp)[((size_t)bb * D_DIM + col) * S_LEN + ss] = f2bf(val);
            } else {
                ((float*)outp)[(size_t)row * D_DIM + col] = val;
            }
        }
    }
}

// ---------------------------------------------------------------- flash attn
// Block = 4 waves = 4 query tiles of the same batch; K/V chunks staged into
// LDS once per block with async-to-LDS DMA, double buffered.
__global__ __launch_bounds__(128, 1) void flash_kernel(
    const unsigned short* __restrict__ Qb,
    const unsigned short* __restrict__ Kb,
    const unsigned short* __restrict__ Vt,
    const float* __restrict__ maskf,
    unsigned short* __restrict__ Ob)
{
    __shared__ __align__(16) unsigned short ldsQ[4][16 * 264];  // 33792 B
    __shared__ __align__(16) unsigned short ldsP[4][16 * 40];   //  5120 B
    __shared__ __align__(16) unsigned short ldsK[2][32 * 264];  // 33792 B
    __shared__ __align__(16) unsigned short ldsV[2][256 * 40];  // 40960 B

    const int tid  = threadIdx.x;
    const int wave = tid >> 5;
    const int t    = tid & 31;
    const int half = t >> 4;
    const int ln   = t & 15;

    const int tile = blockIdx.x * 4 + wave;        // 0..1023
    const int b    = blockIdx.x / (S_LEN / 16 / 4); // uniform across block
    const int i0   = (tile % (S_LEN / 16)) * 16;

    const size_t qbase  = ((size_t)b * S_LEN + i0) * D_DIM;
    const size_t kbbase = (size_t)b * S_LEN * D_DIM;
    const size_t vtbase = (size_t)b * D_DIM * S_LEN;

    unsigned short* Ql = ldsQ[wave];
    unsigned short* Pl = ldsP[wave];

    // Stage this wave's 16x256 Q tile into LDS (padded stride 264).
    {
        const uint4* src = (const uint4*)(Qb + qbase);
#pragma unroll
        for (int it = 0; it < 16; it++) {
            const int idx = it * 32 + t;
            const int r   = idx >> 5;
            const int c8  = (idx & 31) * 8;
            *(uint4*)(Ql + r * 264 + c8) = src[(size_t)r * 32 + (idx & 31)];
        }
    }
    lds_fence();

    // Stage helper: K chunk 32x256 (stride 264), V chunk 256x32 (stride 40).
    auto stage_chunk = [&](int jb, int buf) {
        const unsigned short* Kg = Kb + kbbase + (size_t)jb * D_DIM;
        const unsigned short* Vg = Vt + vtbase + jb;
        unsigned short* lk = ldsK[buf];
        unsigned short* lv = ldsV[buf];
#pragma unroll
        for (int it = 0; it < 8; it++) {
            const int g = it * 128 + tid;            // 1024 uint4 of K
            const int r = g >> 5, c8 = (g & 31) * 8;
            async_copy16(Kg + (size_t)r * D_DIM + c8, lk + r * 264 + c8);
        }
#pragma unroll
        for (int it = 0; it < 8; it++) {
            const int g = it * 128 + tid;            // 1024 uint4 of V
            const int r = g >> 2, c8 = (g & 3) * 8;
            async_copy16(Vg + (size_t)r * S_LEN + c8, lv + r * 40 + c8);
        }
    };

    stage_chunk(0, 0);
    wait_async0();
    __syncthreads();

    const v8f zerov = {0.f,0.f,0.f,0.f,0.f,0.f,0.f,0.f};
    v8f acc[16];
#pragma unroll
    for (int d = 0; d < 16; d++) acc[d] = zerov;

    float rowmax[8], rowsum[8];
#pragma unroll
    for (int r = 0; r < 8; r++) { rowmax[r] = -3.0e38f; rowsum[r] = 0.f; }

    const float scale = 0.0625f; // 1/sqrt(256)

    for (int jc = 0; jc < 128; jc++) {
        const int cur = jc & 1;
        const int jb  = jc * 32;
        if (jc + 1 < 128) stage_chunk(jb + 32, cur ^ 1); // overlap DMA w/ compute

        const unsigned short* lk = ldsK[cur];
        const unsigned short* lv = ldsV[cur];

        // ---- scores: two 16x16 tiles (32 keys), manual reg double-buffer
        v8f s0 = zerov, s1 = zerov;
        ABfrag a, b0, b1, a2, b02, b12;
        a.u[0]  = *(const uint4*)(Ql + ln * 264 + 16 * half);
        a.u[1]  = *(const uint4*)(Ql + ln * 264 + 16 * half + 8);
        b0.u[0] = *(const uint4*)(lk + ln * 264 + 16 * half);
        b0.u[1] = *(const uint4*)(lk + ln * 264 + 16 * half + 8);
        b1.u[0] = *(const uint4*)(lk + (16 + ln) * 264 + 16 * half);
        b1.u[1] = *(const uint4*)(lk + (16 + ln) * 264 + 16 * half + 8);
#pragma unroll
        for (int kk = 0; kk < 8; kk++) {
            if (kk < 7) {
                const int kb = (kk + 1) * 32;
                a2.u[0]  = *(const uint4*)(Ql + ln * 264 + kb + 16 * half);
                a2.u[1]  = *(const uint4*)(Ql + ln * 264 + kb + 16 * half + 8);
                b02.u[0] = *(const uint4*)(lk + ln * 264 + kb + 16 * half);
                b02.u[1] = *(const uint4*)(lk + ln * 264 + kb + 16 * half + 8);
                b12.u[0] = *(const uint4*)(lk + (16 + ln) * 264 + kb + 16 * half);
                b12.u[1] = *(const uint4*)(lk + (16 + ln) * 264 + kb + 16 * half + 8);
            }
            s0 = __builtin_amdgcn_wmma_f32_16x16x32_bf16(
                false, a.v, false, b0.v, (short)0, s0, false, false);
            s1 = __builtin_amdgcn_wmma_f32_16x16x32_bf16(
                false, a.v, false, b1.v, (short)0, s1, false, false);
            if (kk < 7) { a = a2; b0 = b02; b1 = b12; }
        }

        // ---- online softmax (row = r + 8*half; half-wave reductions)
        float alpha[8];
#pragma unroll
        for (int r = 0; r < 8; r++) {
            float mv = fmaxf(s0[r], s1[r]);
            mv = fmaxf(mv, __shfl_xor(mv, 1));
            mv = fmaxf(mv, __shfl_xor(mv, 2));
            mv = fmaxf(mv, __shfl_xor(mv, 4));
            mv = fmaxf(mv, __shfl_xor(mv, 8));
            mv *= scale;
            const float mnew = fmaxf(rowmax[r], mv);
            alpha[r] = __expf(rowmax[r] - mnew);
            rowmax[r] = mnew;
            const float p0 = __expf(s0[r] * scale - mnew);
            const float p1 = __expf(s1[r] * scale - mnew);
            s0[r] = p0; s1[r] = p1;
            float rs = p0 + p1;
            rs += __shfl_xor(rs, 1);
            rs += __shfl_xor(rs, 2);
            rs += __shfl_xor(rs, 4);
            rs += __shfl_xor(rs, 8);
            rowsum[r] = rowsum[r] * alpha[r] + rs;
        }
#pragma unroll
        for (int d = 0; d < 16; d++)
#pragma unroll
            for (int r = 0; r < 8; r++) acc[d][r] *= alpha[r];

        // ---- C-layout -> A-layout transpose of P through LDS (bf16)
#pragma unroll
        for (int r = 0; r < 8; r++) {
            const int rr = r + 8 * half;
            Pl[rr * 40 + ln]      = f2bf(s0[r]);
            Pl[rr * 40 + 16 + ln] = f2bf(s1[r]);
        }
        lds_fence();
        ABfrag pa;
        pa.u[0] = *(const uint4*)(Pl + ln * 40 + 16 * half);
        pa.u[1] = *(const uint4*)(Pl + ln * 40 + 16 * half + 8);

        // ---- O += P(16x32) * V(32x256) from LDS, reg double-buffered
        ABfrag bv, bv2;
        bv.u[0] = *(const uint4*)(lv + ln * 40 + 16 * half);
        bv.u[1] = *(const uint4*)(lv + ln * 40 + 16 * half + 8);
#pragma unroll
        for (int d = 0; d < 16; d++) {
            if (d < 15) {
                const int dn = (d + 1) * 16 + ln;
                bv2.u[0] = *(const uint4*)(lv + dn * 40 + 16 * half);
                bv2.u[1] = *(const uint4*)(lv + dn * 40 + 16 * half + 8);
            }
            acc[d] = __builtin_amdgcn_wmma_f32_16x16x32_bf16(
                false, pa.v, false, bv.v, (short)0, acc[d], false, false);
            if (d < 15) bv = bv2;
        }

        wait_async0();       // next chunk's DMA has landed
        __syncthreads();     // all waves done reading cur before it's reused
    }

    // ---- epilogue: normalize, apply row mask, write bf16 attn output
    float rm[8];
#pragma unroll
    for (int r = 0; r < 8; r++)
        rm[r] = maskf[(size_t)b * S_LEN + i0 + r + 8 * half];
#pragma unroll
    for (int d = 0; d < 16; d++) {
        const int col = d * 16 + ln;
#pragma unroll
        for (int r = 0; r < 8; r++) {
            const int row = i0 + r + 8 * half;
            const float val = acc[d][r] / rowsum[r] * rm[r];
            Ob[((size_t)b * S_LEN + row) * D_DIM + col] = f2bf(val);
        }
    }
}

// ---------------------------------------------------------------- launcher
extern "C" void kernel_launch(void* const* d_in, const int* in_sizes, int n_in,
                              void* d_out, int out_size, void* d_ws, size_t ws_size,
                              hipStream_t stream) {
    (void)in_sizes; (void)n_in; (void)out_size; (void)ws_size;
    const float* src = (const float*)d_in[0];
    const float* Wq  = (const float*)d_in[1];
    const float* bq  = (const float*)d_in[2];
    const float* Wk  = (const float*)d_in[3];
    const float* bk  = (const float*)d_in[4];
    const float* Wv  = (const float*)d_in[5];
    const float* bv  = (const float*)d_in[6];
    const float* Wo  = (const float*)d_in[7];
    const float* bo  = (const float*)d_in[8];
    float* outF = (float*)d_out;

    char* ws = (char*)d_ws;
    size_t off = 0;
    auto alloc = [&](size_t bytes) -> void* {
        void* p = ws + off;
        off = (off + bytes + 255) & ~(size_t)255;
        return p;
    };
    const size_t NE = (size_t)R_ROWS * D_DIM;
    unsigned short* srcb  = (unsigned short*)alloc(NE * 2);
    unsigned short* Wqb   = (unsigned short*)alloc((size_t)D_DIM * D_DIM * 2);
    unsigned short* Wkb   = (unsigned short*)alloc((size_t)D_DIM * D_DIM * 2);
    unsigned short* Wvb   = (unsigned short*)alloc((size_t)D_DIM * D_DIM * 2);
    unsigned short* Wob   = (unsigned short*)alloc((size_t)D_DIM * D_DIM * 2);
    unsigned short* Qb    = (unsigned short*)alloc(NE * 2);
    unsigned short* Kbuf  = (unsigned short*)alloc(NE * 2);
    unsigned short* Vtb   = (unsigned short*)alloc(NE * 2);
    unsigned short* attnb = (unsigned short*)alloc(NE * 2);
    float* maskf          = (float*)alloc((size_t)R_ROWS * 4);

    cvt_bf16_kernel<<<4096, 256, 0, stream>>>(src, srcb, (int)NE);
    cvt_bf16_kernel<<<64, 256, 0, stream>>>(Wq, Wqb, D_DIM * D_DIM);
    cvt_bf16_kernel<<<64, 256, 0, stream>>>(Wk, Wkb, D_DIM * D_DIM);
    cvt_bf16_kernel<<<64, 256, 0, stream>>>(Wv, Wvb, D_DIM * D_DIM);
    cvt_bf16_kernel<<<64, 256, 0, stream>>>(Wo, Wob, D_DIM * D_DIM);

    proj_kernel<0><<<R_ROWS / 16, 128, 0, stream>>>(srcb, Wqb, bq, Qb);
    proj_kernel<0><<<R_ROWS / 16, 128, 0, stream>>>(srcb, Wkb, bk, Kbuf);
    proj_kernel<1><<<R_ROWS / 16, 128, 0, stream>>>(srcb, Wvb, bv, Vtb);

    mask_kernel<<<64, 256, 0, stream>>>(Qb, Kbuf, maskf, R_ROWS);

    flash_kernel<<<R_ROWS / 64, 128, 0, stream>>>(Qb, Kbuf, Vtb, maskf, attnb);

    proj_kernel<2><<<R_ROWS / 16, 128, 0, stream>>>(attnb, Wob, bo, outF);
}
